// RNN_4054449127976
// MI455X (gfx1250) — matrix operand order repaired and checked by reading, not compiled, not run on hardware
//
#include <hip/hip_runtime.h>
#include <cstdint>

#define B_SZ 1024
#define T_SZ 512
#define H_SZ 64
#define G_SZ 256
#define IN_SZ 6

typedef __attribute__((ext_vector_type(16))) _Float16 v16h;
typedef __attribute__((ext_vector_type(8)))  float    v8f;

union F8 { v8f v; float f[8]; };
union AF { v16h h; uint4 u[2]; };

// Branch-free activations: single V_TANH_F32 on gfx1250 if available.
__device__ __forceinline__ float tanh_fast(float x) {
#if __has_builtin(__builtin_amdgcn_tanhf)
  return __builtin_amdgcn_tanhf(x);
#else
  float e = __expf(x + x);                 // tanh(x) = 1 - 2/(e^{2x}+1)
  return 1.0f - 2.0f / (e + 1.0f);
#endif
}
__device__ __forceinline__ float sigf(float x) {
  return 0.5f + 0.5f * tanh_fast(0.5f * x);
}

// ---------------------------------------------------------------------------
// Prep: pad layer-0 input (B,T,6) fp32 -> (B,T,64) f16 (zeros beyond k=5)
// ---------------------------------------------------------------------------
__global__ __launch_bounds__(256) void pad_x_kernel(const float* __restrict__ x,
                                                    _Float16* __restrict__ dst) {
  size_t idx = (size_t)blockIdx.x * 256 + threadIdx.x;   // B*T*64 threads exactly
  size_t row = idx >> 6;
  int j = (int)(idx & 63);
  float v = (j < IN_SZ) ? x[row * IN_SZ + j] : 0.0f;
  dst[idx] = (_Float16)v;
}

// ---------------------------------------------------------------------------
// Prep: swizzle W_ih / W_hh (f32, row-major [256][K]) into WMMA B-fragment
// layout, f16.  Fragment = 16 halves per lane:
//   B 32x16 tile:  n = nt*16 + lane%16 ;  k = kt*32 + (e/8)*16 + (lane/16)*8 + e%8
// dst layout (halves): [mat][nt][kt][lane][e]  -> 32768 halves per layer
// ---------------------------------------------------------------------------
__global__ __launch_bounds__(256) void prep_w_kernel(const float* __restrict__ wih,
                                                     const float* __restrict__ whh,
                                                     int k_ih,
                                                     _Float16* __restrict__ dst) {
  int tid = blockIdx.x * 256 + threadIdx.x;   // 0..32767
  int mat  = tid >> 14;
  int r    = tid & 16383;
  int nt   = r >> 10;
  int kt   = (r >> 9) & 1;
  int lane = (r >> 4) & 31;
  int e    = r & 15;
  int k = kt * 32 + ((e >> 3) << 4) + ((lane >> 4) << 3) + (e & 7);
  int n = nt * 16 + (lane & 15);
  const float* w = mat ? whh : wih;
  int K = mat ? H_SZ : k_ih;
  float val = (k < K) ? w[n * K + k] : 0.0f;
  dst[tid] = (_Float16)val;
}

// Combined bias b_ih + b_hh (256 per layer)
__global__ void prep_b_kernel(const float* __restrict__ bi,
                              const float* __restrict__ bh,
                              float* __restrict__ dst) {
  int g = threadIdx.x;
  dst[g] = bi[g] + bh[g];
}

// ---------------------------------------------------------------------------
// Fused LSTM layer.  Block = 4 waves x 32 lanes, owns 16 batch rows for all
// 512 timesteps.  Wave w handles hidden cols [w*16, w*16+16): gate tiles
// nt = {w, 4+w, 8+w, 12+w} form a complete i/f/g/o quadruple, so the cell
// update is wave-local.  All 16 B-fragments live in VGPRs for the whole
// kernel; per step: 16 v_wmma_f32_16x16x32_f16 per wave, h_t exchanged
// through 2KB LDS with two workgroup barriers.
// ---------------------------------------------------------------------------
__global__ __launch_bounds__(128) void lstm_layer_kernel(
    const _Float16* __restrict__ in,     // [B][T][64] f16
    _Float16* __restrict__ out,          // [B][T][64] f16
    const _Float16* __restrict__ wfrag,  // [2][16][2][32][16] halves
    const float* __restrict__ bias)      // [256] combined
{
  __shared__ _Float16 sH[16 * 64];       // h_t bounce: C-layout -> A-layout

  const int lane = threadIdx.x;          // 0..31
  const int w    = threadIdx.y;          // 0..3 : hidden column tile
  const int tid  = w * 32 + lane;
  const int b0   = blockIdx.x * 16;
  const int hi   = lane >> 4;
  const int lcol = lane & 15;

  // ---- load this wave's 16 weight fragments straight into registers ----
  v16h fw[2][4][2];                      // [mat][gate][kt]
  #pragma unroll
  for (int mat = 0; mat < 2; ++mat)
    #pragma unroll
    for (int gate = 0; gate < 4; ++gate)
      #pragma unroll
      for (int kt = 0; kt < 2; ++kt) {
        const int nt = gate * 4 + w;
        AF a;
        const uint4* p =
            (const uint4*)&wfrag[((((mat * 16 + nt) * 2 + kt) * 32 + lane) << 4)];
        a.u[0] = p[0];
        a.u[1] = p[1];
        fw[mat][gate][kt] = a.h;
      }

  float bcol[4];
  #pragma unroll
  for (int gate = 0; gate < 4; ++gate)
    bcol[gate] = bias[(gate * 4 + w) * 16 + lcol];

  F8 cst;                                // cell state for this wave's 16 cols
  #pragma unroll
  for (int e = 0; e < 8; ++e) cst.f[e] = 0.0f;

  AF ah0, ah1;                           // h_{t-1} A-fragments (K 0..31,32..63)
  uint4 z4; z4.x = z4.y = z4.z = z4.w = 0u;
  ah0.u[0] = z4; ah0.u[1] = z4; ah1.u[0] = z4; ah1.u[1] = z4;

  for (int t = 0; t < T_SZ; ++t) {
    // ---- x_t A-fragments from global (row-major f16 [row][64]) ----
    // A 16x32: lane(<16) K {0..7,16..23}; lane(>=16) K {8..15,24..31}
    AF ax0, ax1;
    {
      size_t rb = ((size_t)(b0 + lcol) * T_SZ + t) * 64 + (size_t)(hi << 3);
      const uint4* p0 = (const uint4*)(in + rb);
      ax0.u[0] = p0[0]; ax0.u[1] = p0[2];
      const uint4* p1 = (const uint4*)(in + rb + 32);
      ax1.u[0] = p1[0]; ax1.u[1] = p1[2];
    }

    // ---- gates for this wave's hidden cols: i, f, g, o ----
    F8 acc[4];
    #pragma unroll
    for (int gate = 0; gate < 4; ++gate) {
      F8 a;
      float bv = bcol[gate];
      #pragma unroll
      for (int e = 0; e < 8; ++e) a.f[e] = bv;
      a.v = __builtin_amdgcn_wmma_f32_16x16x32_f16(
          false, ax0.h, false, fw[0][gate][0], (short)0, a.v, false, false);
      a.v = __builtin_amdgcn_wmma_f32_16x16x32_f16(
          false, ax1.h, false, fw[0][gate][1], (short)0, a.v, false, false);
      a.v = __builtin_amdgcn_wmma_f32_16x16x32_f16(
          false, ah0.h, false, fw[1][gate][0], (short)0, a.v, false, false);
      a.v = __builtin_amdgcn_wmma_f32_16x16x32_f16(
          false, ah1.h, false, fw[1][gate][1], (short)0, a.v, false, false);
      acc[gate] = a;
    }

    // ---- LSTM cell update, elementwise in C-layout (M = e+8*hi, N = lcol) ----
    #pragma unroll
    for (int e = 0; e < 8; ++e) {
      float iv = sigf(acc[0].f[e]);
      float fv = sigf(acc[1].f[e]);
      float gv = tanh_fast(acc[2].f[e]);
      float ov = sigf(acc[3].f[e]);
      float c = fv * cst.f[e] + iv * gv;
      cst.f[e] = c;
      float hv = ov * tanh_fast(c);
      sH[(e + (hi << 3)) * 64 + w * 16 + lcol] = (_Float16)hv;
    }
    __syncthreads();

    // ---- reload full h_t as A-fragments from LDS ----
    {
      int rb = lcol * 64 + (hi << 3);
      const uint4* p0 = (const uint4*)&sH[rb];
      ah0.u[0] = p0[0]; ah0.u[1] = p0[2];
      const uint4* p1 = (const uint4*)&sH[rb + 32];
      ah1.u[0] = p1[0]; ah1.u[1] = p1[2];
    }

    // ---- write h_t to global: 128 threads x one 16B chunk ----
    {
      int r   = tid >> 3;
      int off = (tid & 7) << 3;
      const uint4* s = (const uint4*)&sH[r * 64 + off];
      uint4* d = (uint4*)(out + ((size_t)(b0 + r) * T_SZ + t) * 64 + off);
      d[0] = s[0];
    }
    __syncthreads();                     // protect sH before next step's writes
  }
}

// ---------------------------------------------------------------------------
// MLP head on h2[:, T-1, :]: 64 ->128 relu -> 64 relu -> 2. One block per row.
// ---------------------------------------------------------------------------
__global__ __launch_bounds__(128) void head_kernel(
    const _Float16* __restrict__ h2,   // [B][T][64] f16
    const float* __restrict__ fc1_w, const float* __restrict__ fc1_b,
    const float* __restrict__ fc3_w, const float* __restrict__ fc3_b,
    const float* __restrict__ fc2_w, const float* __restrict__ fc2_b,
    float* __restrict__ outp)
{
  __shared__ float hrow[64];
  __shared__ float z1[128];
  __shared__ float z2[64];
  const int b = blockIdx.x, tid = threadIdx.x;
  if (tid < 64)
    hrow[tid] = (float)h2[((size_t)b * T_SZ + (T_SZ - 1)) * 64 + tid];
  __syncthreads();
  {
    float s = fc1_b[tid];
    #pragma unroll 8
    for (int k = 0; k < 64; ++k) s += hrow[k] * fc1_w[tid * 64 + k];
    z1[tid] = fmaxf(s, 0.0f);
  }
  __syncthreads();
  if (tid < 64) {
    float s = fc3_b[tid];
    #pragma unroll 8
    for (int k = 0; k < 128; ++k) s += z1[k] * fc3_w[tid * 128 + k];
    z2[tid] = fmaxf(s, 0.0f);
  }
  __syncthreads();
  if (tid < 2) {
    float s = fc2_b[tid];
    #pragma unroll 8
    for (int k = 0; k < 64; ++k) s += z2[k] * fc2_w[tid * 64 + k];
    outp[b * 2 + tid] = s;
  }
}

// ---------------------------------------------------------------------------
extern "C" void kernel_launch(void* const* d_in, const int* in_sizes, int n_in,
                              void* d_out, int out_size, void* d_ws, size_t ws_size,
                              hipStream_t stream) {
  (void)in_sizes; (void)n_in; (void)out_size; (void)ws_size;
  const float* x = (const float*)d_in[0];
  const float* w_ih[3] = {(const float*)d_in[1], (const float*)d_in[5], (const float*)d_in[9]};
  const float* w_hh[3] = {(const float*)d_in[2], (const float*)d_in[6], (const float*)d_in[10]};
  const float* b_ih[3] = {(const float*)d_in[3], (const float*)d_in[7], (const float*)d_in[11]};
  const float* b_hh[3] = {(const float*)d_in[4], (const float*)d_in[8], (const float*)d_in[12]};
  const float* fc1_w = (const float*)d_in[13];
  const float* fc1_b = (const float*)d_in[14];
  const float* fc3_w = (const float*)d_in[15];
  const float* fc3_b = (const float*)d_in[16];
  const float* fc2_w = (const float*)d_in[17];
  const float* fc2_b = (const float*)d_in[18];

  char* ws = (char*)d_ws;
  const size_t BUF = (size_t)B_SZ * T_SZ * 64 * sizeof(_Float16);  // 64 MiB
  _Float16* buf0 = (_Float16*)(ws);
  _Float16* buf1 = (_Float16*)(ws + BUF);
  _Float16* buf2 = (_Float16*)(ws + 2 * BUF);
  _Float16* wfrag = (_Float16*)(ws + 3 * BUF);                 // 3 * 64 KiB
  float* bias = (float*)(ws + 3 * BUF + (size_t)3 * 32768 * sizeof(_Float16));

  // layer-0 input: pad to 64-wide f16 so all layers share one code path
  pad_x_kernel<<<dim3((B_SZ * T_SZ * 64) / 256), dim3(256), 0, stream>>>(x, buf0);

  for (int l = 0; l < 3; ++l) {
    prep_w_kernel<<<dim3(128), dim3(256), 0, stream>>>(
        w_ih[l], w_hh[l], (l == 0) ? IN_SZ : H_SZ, wfrag + (size_t)l * 32768);
    prep_b_kernel<<<dim3(1), dim3(256), 0, stream>>>(b_ih[l], b_hh[l], bias + l * 256);
  }

  const _Float16* lin[3] = {buf0, buf1, buf2};
  _Float16* lout[3]      = {buf1, buf2, buf0};   // layer-2 reuses x buffer
  for (int l = 0; l < 3; ++l) {
    lstm_layer_kernel<<<dim3(B_SZ / 16), dim3(32, 4), 0, stream>>>(
        lin[l], lout[l], wfrag + (size_t)l * 32768, bias + l * 256);
  }

  head_kernel<<<dim3(B_SZ), dim3(128), 0, stream>>>(
      buf0, fc1_w, fc1_b, fc3_w, fc3_b, fc2_w, fc2_b, (float*)d_out);
}
